// CustomMultiHeadAttention_8375186227930
// MI455X (gfx1250) — matrix operand order repaired and checked by reading, not compiled, not run on hardware
//
#include <hip/hip_runtime.h>

#define BB 8
#define SS 1024
#define DD 768
#define HH 12
#define DEPTH 64

typedef __attribute__((ext_vector_type(16))) __bf16       v16bf;
typedef __attribute__((ext_vector_type(4)))  unsigned int v4u;
typedef __attribute__((ext_vector_type(4)))  float        v4f;
typedef __attribute__((ext_vector_type(8)))  float        v8f;

union Frag { v16bf bf; v4u q[2]; };

__device__ __forceinline__ unsigned short f2bf(float f) {
    union { float f; unsigned u; } v; v.f = f;
    unsigned r = v.u + 0x7fffu + ((v.u >> 16) & 1u);
    return (unsigned short)(r >> 16);
}

// two 16-byte vector loads -> one 32-byte WMMA operand
__device__ __forceinline__ Frag ld_frag(const unsigned short* p0,
                                        const unsigned short* p1) {
    Frag f;
    f.q[0] = *(const v4u*)p0;
    f.q[1] = *(const v4u*)p1;
    return f;
}

// 8 fp32 (two b128 loads) -> 8 bf16 packed in one 16B chunk
__device__ __forceinline__ v4u pack8v(const float* src) {
    v4f a = *(const v4f*)src;
    v4f b = *(const v4f*)(src + 4);
    v4u pk;
    pk[0] = (unsigned)f2bf(a[0]) | ((unsigned)f2bf(a[1]) << 16);
    pk[1] = (unsigned)f2bf(a[2]) | ((unsigned)f2bf(a[3]) << 16);
    pk[2] = (unsigned)f2bf(b[0]) | ((unsigned)f2bf(b[1]) << 16);
    pk[3] = (unsigned)f2bf(b[2]) | ((unsigned)f2bf(b[3]) << 16);
    return pk;
}

// Async 16B global -> LDS copy (CDNA5 GLOBAL_LOAD_ASYNC_TO_LDS_B128, GVS mode:
// mem = SADDR64 + VADDR32; LDS dest = VGPR byte address). Tracked by ASYNCcnt.
__device__ __forceinline__ void async_cp16(void* lds_dst, const void* base,
                                           unsigned byte_off) {
    unsigned lds_addr = (unsigned)(size_t)lds_dst;
    asm volatile("global_load_async_to_lds_b128 %0, %1, %2"
                 :
                 : "v"(lds_addr), "v"(byte_off), "s"(base)
                 : "memory");
}
__device__ __forceinline__ void wait_async() {
    asm volatile("s_wait_asynccnt 0x0" ::: "memory");
}

// ---------------------------------------------------------------------------
// One-shot fp32 -> bf16 conversion (8 elements / thread, vectorized)
// ---------------------------------------------------------------------------
__global__ void cvt_bf16_kernel(const float* __restrict__ src,
                                unsigned short* __restrict__ dst, int n8) {
    int i = blockIdx.x * 256 + threadIdx.x;
    if (i < n8) *(v4u*)(dst + (size_t)i * 8) = pack8v(src + (size_t)i * 8);
}

// ---------------------------------------------------------------------------
// QKV projection: out[b,h,s,dd] = scale * (Xb @ Wb^T + bias), bf16 head layout
// Panel staging via async global->LDS DMA.
// ---------------------------------------------------------------------------
__global__ __launch_bounds__(128)
void proj_qkv_kernel(const unsigned short* __restrict__ Xb,
                     const unsigned short* __restrict__ Wb,
                     const float* __restrict__ bias,
                     unsigned short* __restrict__ out, float scale) {
    __shared__ __align__(16) unsigned short At[64 * 32];
    __shared__ __align__(16) unsigned short Bt[64 * 32];

    const int tid  = threadIdx.x;
    const int lane = tid & 31;
    const int w    = tid >> 5;
    const int m0   = blockIdx.x * 64;
    const int n0   = blockIdx.y * 64;
    const int rloc = lane & 15;
    const int grp  = (lane >> 4) & 1;

    v8f acc[4];
    for (int t = 0; t < 4; ++t)
        for (int e = 0; e < 8; ++e) acc[t][e] = 0.f;

    for (int k0 = 0; k0 < DD; k0 += 32) {
        __syncthreads();
        for (int ch = tid; ch < 256; ch += 128) {
            int r = ch >> 2, c0 = (ch & 3) * 8;
            async_cp16(At + r * 32 + c0, Xb,
                       (unsigned)(((size_t)(m0 + r) * DD + k0 + c0) * 2));
        }
        for (int ch = tid; ch < 256; ch += 128) {
            int n = ch >> 2, c0 = (ch & 3) * 8;
            async_cp16(Bt + n * 32 + c0, Wb,
                       (unsigned)(((size_t)(n0 + n) * DD + k0 + c0) * 2));
        }
        if (k0 + 32 < DD) {  // prefetch next K-panels
            __builtin_prefetch(Xb + (size_t)(m0 + (tid >> 1)) * DD + k0 + 32, 0, 1);
            __builtin_prefetch(Wb + (size_t)(n0 + (tid >> 1)) * DD + k0 + 32, 0, 1);
        }
        wait_async();
        __syncthreads();

        const unsigned short* ar = At + (w * 16 + rloc) * 32;
        Frag a = ld_frag(ar + grp * 16, ar + grp * 16 + 8);
#pragma unroll
        for (int ns = 0; ns < 4; ++ns) {
            const unsigned short* br = Bt + (ns * 16 + rloc) * 32 + grp * 16;
            Frag b = ld_frag(br, br + 8);
            acc[ns] = __builtin_amdgcn_wmma_f32_16x16x32_bf16(
                false, a.bf, false, b.bf, (short)0, acc[ns], false, false);
        }
    }

    for (int ns = 0; ns < 4; ++ns) {
        int d = n0 + ns * 16 + rloc;
        float bv = bias[d];
        int hh = d >> 6, dd = d & 63;
        for (int vi = 0; vi < 8; ++vi) {
            int rowg = m0 + w * 16 + grp * 8 + vi;
            int b = rowg >> 10, s = rowg & 1023;
            out[(((size_t)b * HH + hh) * SS + s) * DEPTH + dd] =
                f2bf((acc[ns][vi] + bv) * scale);
        }
    }
}

// ---------------------------------------------------------------------------
// Output projection: d_out = AO(bf16) @ Wob^T + bo (fp32 out)
// ---------------------------------------------------------------------------
__global__ __launch_bounds__(128)
void proj_out_kernel(const unsigned short* __restrict__ X,
                     const unsigned short* __restrict__ Wb,
                     const float* __restrict__ bias,
                     float* __restrict__ out) {
    __shared__ __align__(16) unsigned short At[64 * 32];
    __shared__ __align__(16) unsigned short Bt[64 * 32];

    const int tid  = threadIdx.x;
    const int lane = tid & 31;
    const int w    = tid >> 5;
    const int m0   = blockIdx.x * 64;
    const int n0   = blockIdx.y * 64;
    const int rloc = lane & 15;
    const int grp  = (lane >> 4) & 1;

    v8f acc[4];
    for (int t = 0; t < 4; ++t)
        for (int e = 0; e < 8; ++e) acc[t][e] = 0.f;

    for (int k0 = 0; k0 < DD; k0 += 32) {
        __syncthreads();
        for (int ch = tid; ch < 256; ch += 128) {
            int r = ch >> 2, c0 = (ch & 3) * 8;
            async_cp16(At + r * 32 + c0, X,
                       (unsigned)(((size_t)(m0 + r) * DD + k0 + c0) * 2));
        }
        for (int ch = tid; ch < 256; ch += 128) {
            int n = ch >> 2, c0 = (ch & 3) * 8;
            async_cp16(Bt + n * 32 + c0, Wb,
                       (unsigned)(((size_t)(n0 + n) * DD + k0 + c0) * 2));
        }
        if (k0 + 32 < DD) {
            __builtin_prefetch(X + (size_t)(m0 + (tid >> 1)) * DD + k0 + 32, 0, 1);
            __builtin_prefetch(Wb + (size_t)(n0 + (tid >> 1)) * DD + k0 + 32, 0, 1);
        }
        wait_async();
        __syncthreads();

        const unsigned short* ar = At + (w * 16 + rloc) * 32;
        Frag a = ld_frag(ar + grp * 16, ar + grp * 16 + 8);
#pragma unroll
        for (int ns = 0; ns < 4; ++ns) {
            const unsigned short* br = Bt + (ns * 16 + rloc) * 32 + grp * 16;
            Frag b = ld_frag(br, br + 8);
            acc[ns] = __builtin_amdgcn_wmma_f32_16x16x32_bf16(
                false, a.bf, false, b.bf, (short)0, acc[ns], false, false);
        }
    }

    for (int ns = 0; ns < 4; ++ns) {
        int d = n0 + ns * 16 + rloc;
        float bv = bias[d];
        for (int vi = 0; vi < 8; ++vi) {
            int rowg = m0 + w * 16 + grp * 8 + vi;
            out[(size_t)rowg * DD + d] = acc[ns][vi] + bv;
        }
    }
}

// ---------------------------------------------------------------------------
// Flash attention with relative-position bias via the skew identity:
//   Srel[s,t] = q[s] . Er[t + (S-1) - s]   (exact where t <= s)
// ---------------------------------------------------------------------------
__global__ __launch_bounds__(128)
void attn_kernel(const unsigned short* __restrict__ qh,
                 const unsigned short* __restrict__ kh,
                 const unsigned short* __restrict__ vh,
                 const unsigned short* __restrict__ eb,
                 unsigned short* __restrict__ ao) {
    __shared__ __align__(16) unsigned short Qs[64 * 64];
    __shared__ __align__(16) unsigned short Ks[32 * 64];   // [key][depth]
    __shared__ __align__(16) unsigned short Vst[64 * 32];  // [depth][key]
    __shared__ __align__(16) unsigned short Ps[4][16 * 32];
    __shared__ __align__(16) float          Rs[4][16 * 48];

    const int tid  = threadIdx.x;
    const int lane = tid & 31;
    const int w    = tid >> 5;
    const int rloc = lane & 15;
    const int grp  = (lane >> 4) & 1;
    const int s0b  = blockIdx.x * 64;
    const int h    = blockIdx.y;
    const int b    = blockIdx.z;
    const int s0w  = s0b + w * 16;
    const size_t headBase = ((size_t)b * HH + h) * SS * DEPTH;

    for (int i4 = tid; i4 < 512; i4 += 128) {  // Q tile: async DMA
        int r = i4 >> 3, c0 = (i4 & 7) * 8;
        async_cp16(Qs + r * 64 + c0, qh,
                   (unsigned)((headBase + (size_t)(s0b + r) * DEPTH + c0) * 2));
    }
    wait_async();
    __syncthreads();

    const unsigned short* qr = Qs + (w * 16 + rloc) * 64;
    Frag aq0 = ld_frag(qr + grp * 16,      qr + 8 + grp * 16);
    Frag aq1 = ld_frag(qr + 32 + grp * 16, qr + 40 + grp * 16);

    v8f Oacc[4];
    float mrow[8], lrow[8];
    for (int t = 0; t < 4; ++t)
        for (int e = 0; e < 8; ++e) Oacc[t][e] = 0.f;
    for (int vi = 0; vi < 8; ++vi) { mrow[vi] = -1e30f; lrow[vi] = 0.f; }

    const int nT = (s0b + 64) / 32;
    for (int it = 0; it < nT; ++it) {
        const int t0 = it * 32;
        __syncthreads();
        for (int i4 = tid; i4 < 256; i4 += 128) {       // K tile: async DMA
            int kk = i4 >> 3, c0 = (i4 & 7) * 8;
            async_cp16(Ks + kk * 64 + c0, kh,
                       (unsigned)((headBase + (size_t)(t0 + kk) * DEPTH + c0) * 2));
        }
        for (int i4 = tid; i4 < 256; i4 += 128) {       // V: transpose into Vst
            int kk = i4 >> 3, d0 = (i4 & 7) * 8;
            v4u v = *(const v4u*)(vh + headBase + (size_t)(t0 + kk) * DEPTH + d0);
#pragma unroll
            for (int j = 0; j < 4; ++j) {
                Vst[(d0 + 2 * j) * 32 + kk]     = (unsigned short)(v[j] & 0xffffu);
                Vst[(d0 + 2 * j + 1) * 32 + kk] = (unsigned short)(v[j] >> 16);
            }
        }
        if (it + 1 < nT) {  // prefetch next K/V tiles
            __builtin_prefetch(kh + headBase + (size_t)(t0 + 32 + (tid >> 2)) * DEPTH, 0, 1);
            __builtin_prefetch(vh + headBase + (size_t)(t0 + 32 + (tid >> 2)) * DEPTH, 0, 1);
        }
        wait_async();
        __syncthreads();

        if (t0 <= s0w + 15) {  // wave-uniform skip of fully-masked tiles
            const int r0 = t0 + SS - 16 - s0w;

            // --- rel-bias band tiles R[s, r0+j], j in [0,48) ---
            for (int rt = 0; rt < 3; ++rt) {
                v8f racc;
                for (int e = 0; e < 8; ++e) racc[e] = 0.f;
                int rr = r0 + rt * 16 + rloc;
                rr = rr < 0 ? 0 : (rr > SS - 1 ? SS - 1 : rr);
                const unsigned short* er = eb + (size_t)rr * DEPTH + grp * 16;
                Frag be0 = ld_frag(er, er + 8);
                Frag be1 = ld_frag(er + 32, er + 40);
                racc = __builtin_amdgcn_wmma_f32_16x16x32_bf16(
                    false, aq0.bf, false, be0.bf, (short)0, racc, false, false);
                racc = __builtin_amdgcn_wmma_f32_16x16x32_bf16(
                    false, aq1.bf, false, be1.bf, (short)0, racc, false, false);
                for (int vi = 0; vi < 8; ++vi)
                    Rs[w][(grp * 8 + vi) * 48 + rt * 16 + rloc] = racc[vi];
            }
            asm volatile("" ::: "memory");

            // --- scores: two 16-key tiles ---
            float xs[2][8];
            for (int st = 0; st < 2; ++st) {
                v8f sacc;
                for (int e = 0; e < 8; ++e) sacc[e] = 0.f;
                const unsigned short* kr = Ks + (st * 16 + rloc) * 64 + grp * 16;
                Frag bk0 = ld_frag(kr, kr + 8);
                Frag bk1 = ld_frag(kr + 32, kr + 40);
                sacc = __builtin_amdgcn_wmma_f32_16x16x32_bf16(
                    false, aq0.bf, false, bk0.bf, (short)0, sacc, false, false);
                sacc = __builtin_amdgcn_wmma_f32_16x16x32_bf16(
                    false, aq1.bf, false, bk1.bf, (short)0, sacc, false, false);
                for (int vi = 0; vi < 8; ++vi) {
                    int M = grp * 8 + vi;
                    int s = s0w + M;
                    int t = t0 + st * 16 + rloc;
                    float x = sacc[vi] + Rs[w][M * 48 + st * 16 + rloc + 15 - M];
                    if (t > s) x = -1e9f;
                    xs[st][vi] = x;
                }
            }

            // --- online softmax ---
            for (int vi = 0; vi < 8; ++vi) {
                int M = grp * 8 + vi;
                float rm = fmaxf(xs[0][vi], xs[1][vi]);
                for (int d = 1; d < 16; d <<= 1)
                    rm = fmaxf(rm, __shfl_xor(rm, d, 32));
                float mn = fmaxf(mrow[vi], rm);
                float sc = __expf(mrow[vi] - mn);
                float p0 = __expf(xs[0][vi] - mn);
                float p1 = __expf(xs[1][vi] - mn);
                float rs = p0 + p1;
                for (int d = 1; d < 16; d <<= 1)
                    rs += __shfl_xor(rs, d, 32);
                lrow[vi] = lrow[vi] * sc + rs;
                mrow[vi] = mn;
                for (int nd = 0; nd < 4; ++nd) Oacc[nd][vi] *= sc;
                Ps[w][M * 32 + rloc]      = f2bf(p0);
                Ps[w][M * 32 + 16 + rloc] = f2bf(p1);
            }
            asm volatile("" ::: "memory");

            // --- P @ V ---
            const unsigned short* pr = Ps[w] + rloc * 32 + grp * 16;
            Frag ap = ld_frag(pr, pr + 8);
#pragma unroll
            for (int nd = 0; nd < 4; ++nd) {
                const unsigned short* vr = Vst + (nd * 16 + rloc) * 32 + grp * 16;
                Frag bv = ld_frag(vr, vr + 8);
                Oacc[nd] = __builtin_amdgcn_wmma_f32_16x16x32_bf16(
                    false, ap.bf, false, bv.bf, (short)0, Oacc[nd], false, false);
            }
        }
    }

    for (int vi = 0; vi < 8; ++vi) {
        int M = grp * 8 + vi;
        int s = s0w + M;
        float inv = 1.f / lrow[vi];
        for (int nd = 0; nd < 4; ++nd) {
            int d = h * DEPTH + nd * 16 + rloc;
            ao[((size_t)b * SS + s) * DD + d] = f2bf(Oacc[nd][vi] * inv);
        }
    }
}

// ---------------------------------------------------------------------------
extern "C" void kernel_launch(void* const* d_in, const int* in_sizes, int n_in,
                              void* d_out, int out_size, void* d_ws, size_t ws_size,
                              hipStream_t stream) {
    const float* Q  = (const float*)d_in[0];
    const float* K  = (const float*)d_in[1];
    const float* V  = (const float*)d_in[2];
    // d_in[3] = additive causal mask (triu) -> implemented analytically
    const float* Wq = (const float*)d_in[4];
    const float* bq = (const float*)d_in[5];
    const float* Wk = (const float*)d_in[6];
    const float* bk = (const float*)d_in[7];
    const float* Wv = (const float*)d_in[8];
    const float* bv = (const float*)d_in[9];
    const float* Wo = (const float*)d_in[10];
    const float* bo = (const float*)d_in[11];
    const float* Er = (const float*)d_in[12];

    const size_t nX = (size_t)BB * SS * DD;   // 6,291,456 (== B*H*S*DEPTH)
    const size_t nW = (size_t)DD * DD;        //   589,824
    const size_t nE = (size_t)SS * DEPTH;     //    65,536

    unsigned short* p   = (unsigned short*)d_ws;
    unsigned short* Qb  = p; p += nX;
    unsigned short* Kb  = p; p += nX;
    unsigned short* Vb  = p; p += nX;
    unsigned short* Wqb = p; p += nW;
    unsigned short* Wkb = p; p += nW;
    unsigned short* Wvb = p; p += nW;
    unsigned short* Wob = p; p += nW;
    unsigned short* eb  = p; p += nE;
    unsigned short* qh  = p; p += nX;
    unsigned short* kh  = p; p += nX;
    unsigned short* vh  = p; p += nX;
    unsigned short* aoY = p; p += nX;

    // one-shot bf16 conversions
    cvt_bf16_kernel<<<(int)(nX / 8 + 255) / 256, 256, 0, stream>>>(Q, Qb, (int)(nX / 8));
    cvt_bf16_kernel<<<(int)(nX / 8 + 255) / 256, 256, 0, stream>>>(K, Kb, (int)(nX / 8));
    cvt_bf16_kernel<<<(int)(nX / 8 + 255) / 256, 256, 0, stream>>>(V, Vb, (int)(nX / 8));
    cvt_bf16_kernel<<<(int)(nW / 8 + 255) / 256, 256, 0, stream>>>(Wq, Wqb, (int)(nW / 8));
    cvt_bf16_kernel<<<(int)(nW / 8 + 255) / 256, 256, 0, stream>>>(Wk, Wkb, (int)(nW / 8));
    cvt_bf16_kernel<<<(int)(nW / 8 + 255) / 256, 256, 0, stream>>>(Wv, Wvb, (int)(nW / 8));
    cvt_bf16_kernel<<<(int)(nW / 8 + 255) / 256, 256, 0, stream>>>(Wo, Wob, (int)(nW / 8));
    cvt_bf16_kernel<<<(int)(nE / 8 + 255) / 256, 256, 0, stream>>>(Er, eb, (int)(nE / 8));

    dim3 gridP(BB * SS / 64, DD / 64);
    proj_qkv_kernel<<<gridP, 128, 0, stream>>>(Qb, Wqb, bq, qh, 0.125f);  // 1/sqrt(64)
    proj_qkv_kernel<<<gridP, 128, 0, stream>>>(Kb, Wkb, bk, kh, 1.0f);
    proj_qkv_kernel<<<gridP, 128, 0, stream>>>(Vb, Wvb, bv, vh, 1.0f);

    attn_kernel<<<dim3(SS / 64, HH, BB), 128, 0, stream>>>(qh, kh, vh, eb, aoY);

    proj_out_kernel<<<gridP, 128, 0, stream>>>(aoY, Wob, bo, (float*)d_out);
}
